// NLNN_14980845928715
// MI455X (gfx1250) — matrix operand order repaired and checked by reading, not compiled, not run on hardware
//
#include <hip/hip_runtime.h>
#include <math.h>

#define THREADS 256
#define BT 16

namespace {

constexpr int kB = 2048, kS = 512, kI = 128, kH = 256, kO = 64;
constexpr float kEps = 1e-8f;

typedef __bf16 bf16_t;
typedef __attribute__((ext_vector_type(16))) __bf16 v16bf;
typedef __attribute__((ext_vector_type(8)))  __bf16 v8bf;
typedef __attribute__((ext_vector_type(8)))  float  v8f;

// ---- LDS layout (bytes) ----
constexpr unsigned OFF_WIN   = 0;                          // 4*16 tiles * 1KB  = 64KB
constexpr unsigned OFF_WREC  = OFF_WIN  + 4*16*1024;       // 8*16 tiles * 1KB  = 128KB
constexpr unsigned OFF_WOUT  = OFF_WREC + 8*16*1024;       // 8*4  tiles * 1KB  = 32KB
constexpr unsigned OFF_AX    = OFF_WOUT + 8*4*1024;        // [16][128] bf16    = 4KB
constexpr unsigned OFF_AH    = OFF_AX   + 16*128*2;        // [16][256] bf16    = 8KB
constexpr unsigned OFF_HOLD  = OFF_AH   + 16*256*2;        // [16][256] f32     = 16KB
constexpr unsigned OFF_HNEW  = OFF_HOLD + 16*256*4;        // [16][256] f32     = 16KB
constexpr unsigned OFF_ALPHA = OFF_HNEW + 16*256*4;        // [256] f32         = 1KB
constexpr unsigned SMEM_BYTES = OFF_ALPHA + 256*4;         // 275456 B < 320KB

__device__ __forceinline__ v8f wmma_bf16(v16bf a, v16bf b, v8f c) {
  // D = A(16x32 bf16) * B(32x16 bf16) + C(16x16 f32)
  return __builtin_amdgcn_wmma_f32_16x16x32_bf16(false, a, false, b, (short)0, c,
                                                 false, false);
}

// A-fragment from row-major [16][row_halves] bf16 LDS buffer.
// ISA 16-bit A layout: lane m (0-15) + khalf: halves = K[kh*8 .. kh*8+7] ++ K[16+kh*8 .. +7]
// => two contiguous 16B chunks per lane.
__device__ __forceinline__ v16bf load_a_frag(const bf16_t* base, int row_halves,
                                             int kt, int lane) {
  const int m = lane & 15, kh = lane >> 4;
  const bf16_t* p = base + m * row_halves + kt * 32 + kh * 8;
  union { v16bf v; v8bf h[2]; } u;
  u.h[0] = *(const v8bf*)(p);
  u.h[1] = *(const v8bf*)(p + 16);
  return u.v;
}

// Pack W[K][N] (row-major f32) * colscale[n] * extra into bf16 WMMA-B tile layout:
// tile (kt,nt) is 32 lanes x 16 halves contiguous; element (k,n):
//   lane = (n&15) + 16*(((k&31)>>3)&1), half = (k&7) + 8*((k&31)>>4).
__device__ __forceinline__ void pack_weight(const float* __restrict__ W,
                                            bf16_t* __restrict__ dst,
                                            const float* __restrict__ colscale,
                                            float extra, int K, int N, int tid) {
  const int NT = N >> 4;
  for (int e = tid; e < K * N; e += THREADS) {
    const int k = e / N, n = e - k * N;
    float v = W[e] * extra;
    if (colscale) v *= colscale[n];
    const int kt = k >> 5, kk = k & 31;
    const int nt = n >> 4, nl = n & 15;
    const int l = nl + (((kk >> 3) & 1) << 4);
    const int j = (kk & 7) + ((kk >> 4) << 3);
    dst[(unsigned)(kt * NT + nt) * 512u + (unsigned)(l * 16 + j)] = (bf16_t)v;
  }
}

} // namespace

__global__ __launch_bounds__(THREADS, 1)
void nlnn_fwd(const float* __restrict__ x, const float* __restrict__ h_init,
              const float* __restrict__ w_in, const float* __restrict__ w_rec,
              const float* __restrict__ w_out, const float* __restrict__ lam_i,
              const float* __restrict__ lam_r, const float* __restrict__ s_z,
              const float* __restrict__ alpha_raw, float* __restrict__ out) {
  extern __shared__ char smem[];
  bf16_t* sWin   = (bf16_t*)(smem + OFF_WIN);
  bf16_t* sWrec  = (bf16_t*)(smem + OFF_WREC);
  bf16_t* sWout  = (bf16_t*)(smem + OFF_WOUT);
  bf16_t* sAx    = (bf16_t*)(smem + OFF_AX);
  bf16_t* sAh    = (bf16_t*)(smem + OFF_AH);
  float*  sHold  = (float*)(smem + OFF_HOLD);
  float*  sHnew  = (float*)(smem + OFF_HNEW);
  float*  sAlpha = (float*)(smem + OFF_ALPHA);

  const int tid   = threadIdx.x;
  const int wave  = tid >> 5;
  const int lane  = tid & 31;
  const int kh    = lane >> 4;   // which half-wave (C/D row group)
  const int nn    = lane & 15;   // column within N-tile
  const int wgRow = blockIdx.x * BT;

  // ---- one-time: pack weights (lambda/s_z folded in), alpha, h_init ----
  pack_weight(w_in,  sWin,  lam_i,  1.0f,    kI, kH, tid);
  pack_weight(w_rec, sWrec, lam_r,  1.0f,    kH, kH, tid);
  pack_weight(w_out, sWout, nullptr, s_z[0], kH, kO, tid);
  for (int e = tid; e < kH; e += THREADS)
    sAlpha[e] = 1.0f / (1.0f + expf(-alpha_raw[e]));
  for (int e = tid; e < BT * kH; e += THREADS) {
    const int r = e >> 8, c = e & 255;
    const float v = h_init[(unsigned)(wgRow + r) * kH + c];
    sHold[e] = v;                  // e == r*256 + c
    sAh[e]   = (bf16_t)v;
  }
  __syncthreads();

  for (int t = 0; t < kS; ++t) {
    // ---- stage x_t -> bf16 A-layout [16][128] (each thread: 8 floats) ----
    {
      const int e = tid * 8;                 // 256*8 == 16*128 exactly
      const int r = e >> 7, c = e & 127;
      const float* p = x + ((size_t)(wgRow + r) * kS + t) * kI + c;
      float4 f0 = *(const float4*)(p);
      float4 f1 = *(const float4*)(p + 4);
      bf16_t* q = sAx + e;
      q[0] = (bf16_t)f0.x; q[1] = (bf16_t)f0.y; q[2] = (bf16_t)f0.z; q[3] = (bf16_t)f0.w;
      q[4] = (bf16_t)f1.x; q[5] = (bf16_t)f1.y; q[6] = (bf16_t)f1.z; q[7] = (bf16_t)f1.w;
      if (t + 1 < kS) __builtin_prefetch(p + kI, 0, 0);   // next timestep, same row
    }
    __syncthreads();

    // ---- h_pre = x_t @ (Win*li) + h @ (Wrec*lr); wave owns 32 columns ----
    v8f acc0 = {}; v8f acc1 = {};
    const unsigned nt0 = (unsigned)wave * 2u, nt1 = nt0 + 1u;
#pragma unroll
    for (int kt = 0; kt < 4; ++kt) {
      v16bf a  = load_a_frag(sAx, kI, kt, lane);
      v16bf b0 = *(const v16bf*)(sWin + ((unsigned)kt * 16u + nt0) * 512u + (unsigned)lane * 16u);
      v16bf b1 = *(const v16bf*)(sWin + ((unsigned)kt * 16u + nt1) * 512u + (unsigned)lane * 16u);
      acc0 = wmma_bf16(a, b0, acc0);
      acc1 = wmma_bf16(a, b1, acc1);
    }
#pragma unroll
    for (int kt = 0; kt < 8; ++kt) {
      v16bf a  = load_a_frag(sAh, kH, kt, lane);
      v16bf b0 = *(const v16bf*)(sWrec + ((unsigned)kt * 16u + nt0) * 512u + (unsigned)lane * 16u);
      v16bf b1 = *(const v16bf*)(sWrec + ((unsigned)kt * 16u + nt1) * 512u + (unsigned)lane * 16u);
      acc0 = wmma_bf16(a, b0, acc0);
      acc1 = wmma_bf16(a, b1, acc1);
    }
#pragma unroll
    for (int g = 0; g < 8; ++g) {           // C/D layout: row = g + 8*kh, col = nn
      const int m = g + 8 * kh;
      sHnew[m * kH + (int)nt0 * 16 + nn] = acc0[g];
      sHnew[m * kH + (int)nt1 * 16 + nn] = acc1[g];
    }
    __syncthreads();

    // ---- normalize + SLERP (fp32). 16 threads per row; shfl_xor row-reduce ----
    {
      const int r = tid >> 4, sub = tid & 15;
      float hn[16], ho[16];
      float sn = 0.f, so = 0.f, sd = 0.f;
#pragma unroll
      for (int j = 0; j < 16; ++j) {
        const int c = sub + (j << 4);       // stride-16 -> bank-conflict free
        const float a_ = sHnew[r * kH + c];
        const float b_ = sHold[r * kH + c];
        hn[j] = a_; ho[j] = b_;
        sn = fmaf(a_, a_, sn); so = fmaf(b_, b_, so); sd = fmaf(a_, b_, sd);
      }
#pragma unroll
      for (int m = 8; m >= 1; m >>= 1) {    // masks <16: stays inside half-wave == row
        sn += __shfl_xor(sn, m, 32);
        so += __shfl_xor(so, m, 32);
        sd += __shfl_xor(sd, m, 32);
      }
      const float invn = 1.f / (sqrtf(sn) + kEps);
      const float invo = 1.f / (sqrtf(so) + kEps);
      float dot = sd * invn * invo;
      dot = fminf(fmaxf(dot, -1.f + kEps), 1.f - kEps);
      const float theta = acosf(dot);
      const float st = sqrtf(fmaxf(1.f - dot * dot, 0.f));   // sin(theta), theta in [0,pi]
      const bool  use_slerp = st > kEps;
      const float rst = 1.f / (st + kEps);
      float res[16]; float sr = 0.f;
#pragma unroll
      for (int j = 0; j < 16; ++j) {
        const int c = sub + (j << 4);
        const float al = sAlpha[c];
        float sa, ca;
        __sincosf(al * theta, &sa, &ca);    // sin((1-a)t) = st*ca - cos(t)*sa, cos(t)==dot
        const float hnn = hn[j] * invn, hon = ho[j] * invo;
        const float ct = (st * ca - dot * sa) * rst;
        const float cn = sa * rst;
        const float v = use_slerp ? fmaf(ct, hon, cn * hnn) : hnn;
        res[j] = v; sr = fmaf(v, v, sr);
      }
#pragma unroll
      for (int m = 8; m >= 1; m >>= 1) sr += __shfl_xor(sr, m, 32);
      const float invr = 1.f / (sqrtf(sr) + kEps);
#pragma unroll
      for (int j = 0; j < 16; ++j) {
        const int c = sub + (j << 4);
        const float v = res[j] * invr;
        sHold[r * kH + c] = v;
        sAh[r * kH + c]   = (bf16_t)v;
      }
    }
    __syncthreads();

    // ---- out_t = h @ (Wout*s_z): waves 0..3, one 16-col N-tile each ----
    if (wave < 4) {
      v8f acc = {};
#pragma unroll
      for (int kt = 0; kt < 8; ++kt) {
        v16bf a = load_a_frag(sAh, kH, kt, lane);
        v16bf b = *(const v16bf*)(sWout + ((unsigned)(kt * 4 + wave)) * 512u + (unsigned)lane * 16u);
        acc = wmma_bf16(a, b, acc);
      }
#pragma unroll
      for (int g = 0; g < 8; ++g) {
        const int m = g + 8 * kh;
        out[((size_t)(wgRow + m) * kS + t) * kO + wave * 16 + nn] = acc[g];
      }
    }
    __syncthreads();   // protect sAx/sAh/sHnew reuse across iterations
  }

  // ---- h_fin appended after outs ----
  {
    float* hout = out + (size_t)kB * kS * kO;
    for (int e = tid; e < BT * kH; e += THREADS) {
      const int r = e >> 8, c = e & 255;
      hout[(size_t)(wgRow + r) * kH + c] = sHold[e];
    }
  }
}

extern "C" void kernel_launch(void* const* d_in, const int* in_sizes, int n_in,
                              void* d_out, int out_size, void* d_ws, size_t ws_size,
                              hipStream_t stream) {
  (void)in_sizes; (void)n_in; (void)out_size; (void)d_ws; (void)ws_size;
  const float* x         = (const float*)d_in[0];
  const float* h_init    = (const float*)d_in[1];
  const float* w_in      = (const float*)d_in[2];
  const float* w_rec     = (const float*)d_in[3];
  const float* w_out     = (const float*)d_in[4];
  const float* lam_i     = (const float*)d_in[5];
  const float* lam_r     = (const float*)d_in[6];
  const float* s_z       = (const float*)d_in[7];
  const float* alpha_raw = (const float*)d_in[8];
  float* out = (float*)d_out;

  hipFuncSetAttribute(reinterpret_cast<const void*>(&nlnn_fwd),
                      hipFuncAttributeMaxDynamicSharedMemorySize, (int)SMEM_BYTES);

  nlnn_fwd<<<dim3(kB / BT), dim3(THREADS), SMEM_BYTES, stream>>>(
      x, h_init, w_in, w_rec, w_out, lam_i, lam_r, s_z, alpha_raw, out);
}
